// RNNModel_90056874262846
// MI455X (gfx1250) — matrix operand order repaired and checked by reading, not compiled, not run on hardware
//
#include <hip/hip_runtime.h>
#include <hip/hip_bf16.h>

// ---------------------------------------------------------------------------
// RNN LM: emb gather -> 6x (input-proj GEMM + sequential Elman scan) -> logits
// Matmuls on v_wmma_f32_16x16x32_f16; GEMM tiles staged through LDS with the
// gfx1250 async-to-LDS path (global_load_async_to_lds_b128 + s_wait_asynccnt).
// ---------------------------------------------------------------------------

typedef __attribute__((ext_vector_type(16))) _Float16 v16h;
typedef __attribute__((ext_vector_type(8)))  _Float16 v8h;
typedef __attribute__((ext_vector_type(8)))  float    v8f;

#define VOCAB  32000
#define EMBED  256
#define HIDDEN 384
#define LAYERS 6
#define BATCH  16
#define SEQ    512
#define SB     (BATCH * SEQ)   // 8192 rows

// ---- gfx1250 async copy: global -> LDS, 16B per lane, tracked by ASYNCcnt --
__device__ __forceinline__ void async_copy_b128(const _Float16* gsrc, _Float16* ldst) {
    // LDS aperture: generic addr[31:0] IS the LDS byte offset.
    unsigned loff = (unsigned)(uintptr_t)ldst;
    unsigned long long g = (unsigned long long)(uintptr_t)gsrc;
    asm volatile("global_load_async_to_lds_b128 %0, %1, off"
                 :: "v"(loff), "v"(g) : "memory");
}
#define S_WAIT_ASYNCCNT_0() asm volatile("s_wait_asynccnt 0x0" ::: "memory")
#define S_WAIT_ASYNCCNT_4() asm volatile("s_wait_asynccnt 0x4" ::: "memory")

// ---- fragment loaders (wave32 layouts per CDNA5 ISA 7.12.2) ----------------
// A (16x32 MxK, 16-bit): lane<16 holds M=lane, K=0..7 (h0..7) / 16..23 (h8..15);
//                        lane>=16 holds M=lane-16, K=8..15 / 24..31.
__device__ __forceinline__ v16h load_a_frag(const _Float16* __restrict__ base,
                                            int row, int ld, int k0, int lane) {
    const _Float16* p = base + (size_t)row * ld + k0 + ((lane >> 4) << 3);
    v8h lo = *(const v8h*)p;
    v8h hi = *(const v8h*)(p + 16);
    return __builtin_shufflevector(lo, hi, 0,1,2,3,4,5,6,7,8,9,10,11,12,13,14,15);
}

// B (32x16 KxN, 16-bit), sourced from row-major W[N][K] (so B = W^T):
// lane<16 holds N=lane, K=0..7 / 16..23; lane>=16 holds N=lane-16, K=8..15 / 24..31.
__device__ __forceinline__ v16h load_b_frag(const _Float16* __restrict__ Wbase,
                                            int nrow, int ld, int k0, int lane) {
    const _Float16* p = Wbase + (size_t)nrow * ld + k0 + ((lane >> 4) << 3);
    v8h lo = *(const v8h*)p;
    v8h hi = *(const v8h*)(p + 16);
    return __builtin_shufflevector(lo, hi, 0,1,2,3,4,5,6,7,8,9,10,11,12,13,14,15);
}

// ---- embedding gather: out[s][b][e] = (f16) emb[x[b][s]][e] ----------------
__global__ __launch_bounds__(256) void embed_gather(const int* __restrict__ x,
                                                    const float* __restrict__ emb,
                                                    _Float16* __restrict__ out) {
    int tid = blockIdx.x * 256 + threadIdx.x;      // S*B*E threads
    int e = tid & (EMBED - 1);
    int b = (tid >> 8) & (BATCH - 1);
    int s = tid >> 12;
    int tok = x[b * SEQ + s];
    out[tid] = (_Float16)emb[(size_t)tok * EMBED + e];
}

// ---- f32 -> f16 conversion -------------------------------------------------
__global__ __launch_bounds__(256) void f32_to_f16(const float* __restrict__ src,
                                                  _Float16* __restrict__ dst, int n) {
    int i = blockIdx.x * 256 + threadIdx.x;
    if (i < n) dst[i] = (_Float16)src[i];
}

// ---- WMMA GEMM with async-LDS double buffering -----------------------------
// C[M,N] = A[M,K] * W[N,K]^T + bias[N].
// 256 threads = 8 waves; block tile 128x128; wave tile 64x32 (4x2 wmma tiles).
// Per k-step: stage 128x32 A-tile + 128x32 W-tile (8KB each) via async-to-LDS.
// perm_bs: remap output row m = s*16+b -> b*512+s (final logits are [B,S,V]).
__global__ __launch_bounds__(256) void gemm_f16f32(const _Float16* __restrict__ A,
                                                   const _Float16* __restrict__ W,
                                                   const float* __restrict__ bias,
                                                   float* __restrict__ C,
                                                   int M, int N, int K, int perm_bs) {
    __shared__ _Float16 smA[2][128 * 32];
    __shared__ _Float16 smW[2][128 * 32];

    const int tid  = threadIdx.x;
    const int lane = tid & 31;
    const int wave = tid >> 5;
    const int lrow = lane & 15;
    const int lhi  = lane >> 4;
    const int mblk0 = blockIdx.x * 128;
    const int nblk0 = blockIdx.y * 128;
    const int wm = (wave & 1) * 64;    // wave M offset within block
    const int wn = (wave >> 1) * 32;   // wave N offset within block

    // Stage one 128x32 k-slice of A and W into LDS buffer `buf`.
    // 512 16B chunks per tile, 2 per thread per tile -> 4 async ops / thread.
    auto stage = [&](int kq, int buf) {
#pragma unroll
        for (int c = 0; c < 2; ++c) {
            int chunk = tid + (c << 8);          // 0..511
            int row = chunk >> 2;
            int col = (chunk & 3) << 3;          // 0,8,16,24 halfs
            async_copy_b128(A + (size_t)(mblk0 + row) * K + kq + col,
                            &smA[buf][row * 32 + col]);
            async_copy_b128(W + (size_t)(nblk0 + row) * K + kq + col,
                            &smW[buf][row * 32 + col]);
        }
    };

    v8f acc[4][2] = {};
    const int nk = K >> 5;

    stage(0, 0);
    for (int kk = 0; kk < nk; ++kk) {
        const int cur = kk & 1;
        if (kk + 1 < nk) {
            stage((kk + 1) << 5, cur ^ 1);
            S_WAIT_ASYNCCNT_4();   // 4 in flight for next stage; current done
        } else {
            S_WAIT_ASYNCCNT_0();
        }
        __syncthreads();           // staged tile visible to all waves

        v16h a[4], b[2];
#pragma unroll
        for (int i = 0; i < 4; ++i)
            a[i] = load_a_frag(smA[cur], wm + i * 16 + lrow, 32, 0, lane);
#pragma unroll
        for (int j = 0; j < 2; ++j)
            b[j] = load_b_frag(smW[cur], wn + j * 16 + lrow, 32, 0, lane);
#pragma unroll
        for (int i = 0; i < 4; ++i)
#pragma unroll
            for (int j = 0; j < 2; ++j)
                acc[i][j] = __builtin_amdgcn_wmma_f32_16x16x32_f16(
                    false, a[i], false, b[j], (short)0, acc[i][j], false, false);

        __syncthreads();           // reads done before this buffer is restaged
    }

#pragma unroll
    for (int j = 0; j < 2; ++j) {
        const int col = nblk0 + wn + j * 16 + lrow;
        const float bj = bias[col];
#pragma unroll
        for (int i = 0; i < 4; ++i) {
#pragma unroll
            for (int v = 0; v < 8; ++v) {
                int row = mblk0 + wm + i * 16 + v + 8 * lhi;
                int orow = perm_bs ? ((row & (BATCH - 1)) * SEQ + (row >> 4)) : row;
                C[(size_t)orow * N + col] = acc[i][j][v] + bj;
            }
        }
    }
}

// ---- sequential Elman scan for one layer -----------------------------------
// One workgroup (8 waves). h kept as f16 in LDS (16x384 = 12KB).
// h_t = tanh(pre_t + h_{t-1} @ Wh^T + bh);  writes f16 activations [S,B,H].
__global__ __launch_bounds__(256) void rnn_scan(const float* __restrict__ pre,
                                                const _Float16* __restrict__ Wh,
                                                const float* __restrict__ bh,
                                                _Float16* __restrict__ out16,
                                                int S) {
    __shared__ _Float16 h16[BATCH * HIDDEN];

    const int tid  = threadIdx.x;
    const int lane = tid & 31;
    const int wave = tid >> 5;
    const int lrow = lane & 15;
    const int lhi  = lane >> 4;

    for (int i = tid; i < BATCH * HIDDEN; i += 256) h16[i] = (_Float16)0.f;
    __syncthreads();

    // each wave owns 3 N-tiles of 16: n0 = (wave*3 + nt)*16
    float bcol[3];
#pragma unroll
    for (int nt = 0; nt < 3; ++nt) bcol[nt] = bh[(wave * 3 + nt) * 16 + lrow];

    for (int t = 0; t < S; ++t) {
        v8f acc[3] = {};
        for (int k0 = 0; k0 < HIDDEN; k0 += 32) {
            v16h a = load_a_frag(h16, lrow, HIDDEN, k0, lane);   // rows = batch
#pragma unroll
            for (int nt = 0; nt < 3; ++nt) {
                v16h b = load_b_frag(Wh, (wave * 3 + nt) * 16 + lrow, HIDDEN, k0, lane);
                acc[nt] = __builtin_amdgcn_wmma_f32_16x16x32_f16(
                    false, a, false, b, (short)0, acc[nt], false, false);
            }
        }
        __syncthreads();   // all waves done reading h16 for this step

        const float* pt = pre + (size_t)t * BATCH * HIDDEN;
        _Float16* ot = out16 + (size_t)t * BATCH * HIDDEN;
#pragma unroll
        for (int nt = 0; nt < 3; ++nt) {
            const int col = (wave * 3 + nt) * 16 + lrow;
#pragma unroll
            for (int v = 0; v < 8; ++v) {
                int b = v + 8 * lhi;   // batch row
                float val = acc[nt][v] + pt[b * HIDDEN + col] + bcol[nt];
                _Float16 hh = (_Float16)tanhf(val);
                h16[b * HIDDEN + col] = hh;
                ot[b * HIDDEN + col] = hh;
            }
        }
        __syncthreads();   // h16 fully updated before next step reads it
    }
}

// ---------------------------------------------------------------------------
extern "C" void kernel_launch(void* const* d_in, const int* in_sizes, int n_in,
                              void* d_out, int out_size, void* d_ws, size_t ws_size,
                              hipStream_t stream) {
    (void)in_sizes; (void)n_in; (void)out_size; (void)ws_size;

    const int*   x     = (const int*)d_in[0];
    const float* emb   = (const float*)d_in[1];
    const float* W_ih0 = (const float*)d_in[2];
    const float* W_ih  = (const float*)d_in[3];
    const float* W_hh  = (const float*)d_in[4];
    const float* b_ih  = (const float*)d_in[5];
    const float* b_hh  = (const float*)d_in[6];
    const float* W_out = (const float*)d_in[7];
    const float* b_out = (const float*)d_in[8];
    float* out = (float*)d_out;

    char* ws = (char*)d_ws;
    size_t off = 0;
    auto carve = [&](size_t bytes) {
        char* p = ws + off;
        off += (bytes + 255) & ~(size_t)255;
        return p;
    };
    _Float16* actA   = (_Float16*)carve((size_t)SB * HIDDEN * 2);
    _Float16* actB   = (_Float16*)carve((size_t)SB * HIDDEN * 2);
    float*    pre    = (float*)   carve((size_t)SB * HIDDEN * 4);
    _Float16* wi0_16 = (_Float16*)carve((size_t)HIDDEN * EMBED * 2);
    _Float16* wih_16 = (_Float16*)carve((size_t)(LAYERS - 1) * HIDDEN * HIDDEN * 2);
    _Float16* whh_16 = (_Float16*)carve((size_t)LAYERS * HIDDEN * HIDDEN * 2);
    _Float16* wout_16= (_Float16*)carve((size_t)VOCAB * HIDDEN * 2);

    // 1. embedding gather (-> f16 [S,B,E])
    embed_gather<<<(SB * EMBED) / 256, 256, 0, stream>>>(x, emb, actA);

    // 2. weight conversions to f16
    auto cvt = [&](const float* s, _Float16* d, int n) {
        f32_to_f16<<<(n + 255) / 256, 256, 0, stream>>>(s, d, n);
    };
    cvt(W_ih0,  wi0_16,  HIDDEN * EMBED);
    cvt(W_ih,   wih_16,  (LAYERS - 1) * HIDDEN * HIDDEN);
    cvt(W_hh,   whh_16,  LAYERS * HIDDEN * HIDDEN);
    cvt(W_out,  wout_16, VOCAB * HIDDEN);

    // 3. layers: input-proj GEMM (parallel) + scan (sequential)
    _Float16* act_in = actA;
    _Float16* act_out = actB;
    for (int l = 0; l < LAYERS; ++l) {
        const _Float16* Wi = (l == 0) ? wi0_16 : (wih_16 + (size_t)(l - 1) * HIDDEN * HIDDEN);
        const int K = (l == 0) ? EMBED : HIDDEN;
        dim3 grid(SB / 128, HIDDEN / 128);
        gemm_f16f32<<<grid, 256, 0, stream>>>(act_in, Wi, b_ih + l * HIDDEN, pre,
                                              SB, HIDDEN, K, 0);
        rnn_scan<<<1, 256, 0, stream>>>(pre, whh_16 + (size_t)l * HIDDEN * HIDDEN,
                                        b_hh + l * HIDDEN, act_out, SEQ);
        _Float16* t = act_in; act_in = act_out; act_out = t;
    }

    // 4. logits: [8192,384] @ [384,32000]^T + b_out, rows permuted to [B,S]
    dim3 gridO(SB / 128, VOCAB / 128);
    gemm_f16f32<<<gridO, 256, 0, stream>>>(act_in, wout_16, b_out, out,
                                           SB, VOCAB, HIDDEN, 1);
}
